// TemporalConvBlock_68410239091351
// MI455X (gfx1250) — matrix operand order repaired and checked by reading, not compiled, not run on hardware
//
#include <hip/hip_runtime.h>
#include <hip/hip_bf16.h>
#include <math.h>

// ---------------------------------------------------------------------------
// ConvLSTM2D x2 on MI455X (gfx1250).
// Implicit-GEMM conv on v_wmma_f32_16x16x32_bf16, LSTM gates fused in-kernel.
// B=4 T=16 H=W=64 Cin=F=32, k=3, gate width 4F=128.
//
// Per layer, per timestep, ONE kernel:
//   z = conv(x_t, Wx) + conv(h_{t-1}, Wh)          (18 K=32 WMMA chunks/wave)
//   -> accumulators staged to LDS -> gate math -> c update
//   -> h written as bf16 into this layer's hidden-sequence tensor
//      (layer 2 additionally writes f32 h into d_out)
// Hidden sequences live in workspace as bf16 (lossless vs. feeding the WMMA,
// which consumes bf16 anyway), so no buffer is ever read and written by the
// same launch.
// ---------------------------------------------------------------------------

typedef __bf16 bf16_t;
typedef __attribute__((ext_vector_type(16))) __bf16 v16bf;
typedef __attribute__((ext_vector_type(8)))  __bf16 v8bf;
typedef __attribute__((ext_vector_type(8)))  float  v8f;

#define BB   4
#define TT   16
#define HH   64
#define WW   64
#define CC   32      // Cin == F
#define G4   128     // 4*F gate channels
#define NPIX (BB*HH*WW)          // 16384 pixels per timestep
#define HALO (3*34*32)           // 3 rows x 34 cols x 32 ch staged in LDS
#define SMEM_BYTES (2*HALO*2)    // 13056 B: two bf16 halo tiles; < 32*128*4 z-tile

__device__ __forceinline__ bf16_t f2bf(float f) {
  unsigned u = __float_as_uint(f);
  u = (u + 0x7FFFu + ((u >> 16) & 1u)) >> 16;     // round-to-nearest-even
  unsigned short s = (unsigned short)u;
  bf16_t r;
  __builtin_memcpy(&r, &s, 2);
  return r;
}

// ---------------------------------------------------------------------------
// Bake f32 HWIO weights (3,3,32,128) into the per-lane B-fragment layout for
// v_wmma_f32_16x16x32_bf16: lane L -> column n = L%16, K = 8*(L/16)+e (e<8), 16+8*(L/16)+(e-8) (e>=8).
// dst layout: [tap r(9)][n-chunk(8)][lane(32)][e(16)] = 36864 bf16 per conv.
// ---------------------------------------------------------------------------
__global__ void prep_weights_kernel(const float* __restrict__ w,
                                    bf16_t* __restrict__ dst) {
  int idx  = blockIdx.x * 256 + threadIdx.x;      // < 36864
  int e    = idx & 15;
  int lane = (idx >> 4) & 31;
  int nc   = (idx >> 9) & 7;
  int r    = idx >> 12;                           // tap 0..8
  int k    = (e < 8) ? 8 * (lane >> 4) + e : 16 + 8 * (lane >> 4) + (e - 8); // input channel
  int n    = nc * 16 + (lane & 15);               // gate channel
  dst[idx] = f2bf(w[(r * CC + k) * G4 + n]);
}

// ---------------------------------------------------------------------------
// Fused ConvLSTM step. Workgroup = 256 threads (8 waves).
// M-tile = 32 consecutive pixels (half an image row); wave w owns gate
// columns [16w,16w+16); two 16x16 f32 accumulators per wave.
// ---------------------------------------------------------------------------
__global__ __launch_bounds__(256) void convlstm_step_kernel(
    const float*  __restrict__ xf,    // f32 input (B,T,H,W,C), or null
    const bf16_t* __restrict__ xb,    // bf16 input (B,T,H,W,C), or null
    bf16_t*                    hseq,  // this layer's hidden seq (B,T,H,W,F) bf16
    int t,
    const bf16_t* __restrict__ wX,    // baked input-conv weights
    const bf16_t* __restrict__ wH,    // baked recurrent-conv weights
    const float*  __restrict__ bias,  // (128)
    float*        __restrict__ cst,   // (NPIX,F) cell state
    float*        __restrict__ outf)  // d_out (B,T,H,W,F) f32, or null
{
  __shared__ __align__(16) char smem[SMEM_BYTES];
  bf16_t* lx = (bf16_t*)smem;         // [3][34][32]
  bf16_t* lh = lx + HALO;             // [3][34][32]

  const int tid   = threadIdx.x;
  const int pbase = blockIdx.x * 32;          // first pixel of M-tile
  const int b     = pbase >> 12;
  const int y0    = (pbase >> 6) & 63;
  const int x0    = pbase & 63;               // 0 or 32

  const size_t fx = (size_t)(b * TT + t) * (HH * WW * CC);       // x frame
  const size_t fh = (size_t)(b * TT + (t - 1)) * (HH * WW * CC); // h_{t-1} frame

  // ---- stage 3x34x32 halos for both GEMM operands ----
  const bf16_t zb = f2bf(0.0f);
  for (int i = tid; i < HALO; i += 256) {
    int c = i & 31;
    int j = (i >> 5) % 34;
    int r = i / (34 * 32);
    int yy = y0 + r - 1;
    int xx = x0 + j - 1;
    bool ok = (yy >= 0) && (yy < HH) && (xx >= 0) && (xx < WW);
    int off = (yy * WW + xx) * CC + c;
    bf16_t xv = zb, hv = zb;
    if (ok) xv = xf ? f2bf(xf[fx + off]) : xb[fx + off];
    if (ok && t > 0) hv = hseq[fh + off];
    lx[i] = xv;
    lh[i] = hv;
  }
  __syncthreads();

  const int wv = tid >> 5;        // wave id = gate n-chunk
  const int L  = tid & 31;        // lane
  const int g  = L >> 4;
  const int nl = L & 15;

  v8f acc0 = {};                  // pixels pbase+0..15
  v8f acc1 = {};                  // pixels pbase+16..31

  // 9 input taps + 9 recurrent taps; each tap = one K=32 bf16 WMMA chunk.
  #pragma unroll
  for (int cc2 = 0; cc2 < 18; ++cc2) {
    const int r  = (cc2 < 9) ? cc2 : cc2 - 9;
    const bf16_t* lds = (cc2 < 9) ? lx : lh;
    const bf16_t* wb  = (cc2 < 9) ? wX : wH;
    const int ky = r / 3, kx = r % 3;
    // A-fragment (16-bit A 16x32): lane = row M%16, K in {8g..8g+7} U {16+8g..}
    int base0 = (ky * 34 + (nl + kx)) * 32 + 8 * g;
    int base1 = (ky * 34 + (nl + 16 + kx)) * 32 + 8 * g;
    v8bf a0lo = *(const v8bf*)&lds[base0];
    v8bf a0hi = *(const v8bf*)&lds[base0 + 16];
    v8bf a1lo = *(const v8bf*)&lds[base1];
    v8bf a1hi = *(const v8bf*)&lds[base1 + 16];
    v16bf a0 = __builtin_shufflevector(a0lo, a0hi, 0,1,2,3,4,5,6,7,8,9,10,11,12,13,14,15);
    v16bf a1 = __builtin_shufflevector(a1lo, a1hi, 0,1,2,3,4,5,6,7,8,9,10,11,12,13,14,15);
    v16bf bw = *(const v16bf*)&wb[(((r * 8) + wv) * 32 + L) * 16];
    acc0 = __builtin_amdgcn_wmma_f32_16x16x32_bf16(false, a0, false, bw,
                                                   (short)0, acc0, false, false);
    acc1 = __builtin_amdgcn_wmma_f32_16x16x32_bf16(false, a1, false, bw,
                                                   (short)0, acc1, false, false);
  }

  // ---- fused LSTM epilogue: stage z tile (32 pix x 128 gates) in LDS ----
  __syncthreads();                       // all waves done reading halos
  float* zl = (float*)smem;              // reuse halo LDS as z tile
  {
    const int col = wv * 16 + nl;        // C/D layout: vgpr i -> M=i+8g, N=nl
    #pragma unroll
    for (int i = 0; i < 8; ++i) {
      int m = i + 8 * g;
      zl[m * G4 + col]        = acc0[i];
      zl[(16 + m) * G4 + col] = acc1[i];
    }
  }
  __syncthreads();

  // 32 pixels x 32 features = 1024 elements; 4 per thread.
  #pragma unroll
  for (int k = 0; k < 4; ++k) {
    int e  = tid + k * 256;
    int pl = e >> 5;
    int f0 = e & 31;
    const float* zr = zl + pl * G4;
    float zi = zr[f0]      + bias[f0];
    float zf = zr[32 + f0] + bias[32 + f0];
    float zc = zr[64 + f0] + bias[64 + f0];
    float zo = zr[96 + f0] + bias[96 + f0];
    float ig = 1.0f / (1.0f + __expf(-zi));
    float fg = 1.0f / (1.0f + __expf(-zf));
    int p  = pbase + pl;
    int ci = p * CC + f0;
    float cn = fg * cst[ci] + ig * tanhf(zc);
    float og = 1.0f / (1.0f + __expf(-zo));
    float h  = og * tanhf(cn);
    cst[ci] = cn;
    int bb = p >> 12, sp = p & 4095;
    size_t o = (size_t)((bb * TT + t) * 4096 + sp) * CC + f0;
    hseq[o] = f2bf(h);
    if (outf) outf[o] = h;
  }
}

// ---------------------------------------------------------------------------
// Host orchestration. Inputs: x, k1, rk1, b1, k2, rk2, b2 (all f32).
// Workspace: baked weights (0.3MB) + hseq1 (16MB) + hseq2 (16MB) + c (2MB).
// ---------------------------------------------------------------------------
extern "C" void kernel_launch(void* const* d_in, const int* in_sizes, int n_in,
                              void* d_out, int out_size, void* d_ws, size_t ws_size,
                              hipStream_t stream) {
  const float* x   = (const float*)d_in[0];
  const float* k1  = (const float*)d_in[1];
  const float* rk1 = (const float*)d_in[2];
  const float* b1  = (const float*)d_in[3];
  const float* k2  = (const float*)d_in[4];
  const float* rk2 = (const float*)d_in[5];
  const float* b2  = (const float*)d_in[6];
  float* out = (float*)d_out;

  char* ws = (char*)d_ws;
  const size_t WSZ  = 9 * 8 * 32 * 16;                 // 36864 bf16 per conv
  const size_t HSEQ = (size_t)BB * TT * HH * WW * CC;  // 8388608 elements
  bf16_t* wB = (bf16_t*)ws;
  size_t off = ((4 * WSZ * sizeof(bf16_t)) + 255) & ~(size_t)255;
  bf16_t* hseq1 = (bf16_t*)(ws + off); off += HSEQ * sizeof(bf16_t);  // 16MB
  bf16_t* hseq2 = (bf16_t*)(ws + off); off += HSEQ * sizeof(bf16_t);  // 16MB
  float*  cst   = (float*)(ws + off);                                  // 2MB

  prep_weights_kernel<<<144, 256, 0, stream>>>(k1,  wB + 0 * WSZ);
  prep_weights_kernel<<<144, 256, 0, stream>>>(rk1, wB + 1 * WSZ);
  prep_weights_kernel<<<144, 256, 0, stream>>>(k2,  wB + 2 * WSZ);
  prep_weights_kernel<<<144, 256, 0, stream>>>(rk2, wB + 3 * WSZ);

  // Layer 1: f32 x input -> bf16 hidden sequence hseq1.
  hipMemsetAsync(cst, 0, (size_t)NPIX * CC * sizeof(float), stream);
  for (int t = 0; t < TT; ++t) {
    convlstm_step_kernel<<<NPIX / 32, 256, 0, stream>>>(
        x, nullptr, hseq1, t, wB + 0 * WSZ, wB + 1 * WSZ, b1, cst, nullptr);
  }

  // Layer 2: bf16 hseq1 input -> hseq2 (recurrence) + f32 d_out.
  hipMemsetAsync(cst, 0, (size_t)NPIX * CC * sizeof(float), stream);
  for (int t = 0; t < TT; ++t) {
    convlstm_step_kernel<<<NPIX / 32, 256, 0, stream>>>(
        nullptr, hseq1, hseq2, t, wB + 2 * WSZ, wB + 3 * WSZ, b2, cst, out);
  }
}
